// FMoETransformerMLP_25503515804133
// MI455X (gfx1250) — matrix operand order repaired and checked by reading, not compiled, not run on hardware
//
#include <hip/hip_runtime.h>
#include <hip/hip_bf16.h>

// ---------------------------------------------------------------------------
// MoE transformer MLP for MI455X (gfx1250, wave32, WMMA).
//   x:[1,8192,1024] f32, 8 experts, top-2 router, gate = 0.5 per chosen expert.
//   out[t] = sum_{e in top2(t)} 0.5 * MLP_e(x[t])
// Strategy: route first (device-side lists), then run only the assigned
// token-expert pairs through a fully fused bf16-WMMA MLP, 16-token tiles,
// with x/h1/h2 resident in the 320KB LDS. Weights converted f32->bf16 and
// transposed once per launch so B operands are contiguous b128 loads.
// Workspace: 256B counters + 256KB lists + 384MB bf16 transposed weights.
// ---------------------------------------------------------------------------

#define D_MODEL 1024
#define D_HID   4096
#define N_EXP   8
#define N_TOK   8192
#define TILE_M  16
#define XS_STRIDE (D_MODEL + 8)   // +8 bf16 = +16B row pad -> LDS bank spread
#define H_STRIDE  (D_HID + 8)

typedef __attribute__((ext_vector_type(16))) __bf16 v16bf;
typedef __attribute__((ext_vector_type(8)))  float  v8f;

union BfPack { v16bf v; uint4 u[2]; };

__device__ __forceinline__ unsigned short f32_to_bf16(float f) {
  unsigned u = __float_as_uint(f);
  u += 0x7fffu + ((u >> 16) & 1u);           // round-to-nearest-even
  return (unsigned short)(u >> 16);
}

__device__ __forceinline__ float gelu_exact(float v) {
  return 0.5f * v * (1.0f + erff(v * 0.70710678118654752440f));
}

// A operand (16x32 bf16, ISA 7.12.2): lane row = lane&15; lanes 0-15 hold
// K {0..7, 16..23}, lanes 16-31 hold K {8..15, 24..31}. Two b128 loads.
__device__ __forceinline__ v16bf load_opA(const unsigned short* row, int k0, int lane) {
  const unsigned short* p = row + k0 + (((lane >> 4) & 1) << 3);
  BfPack r;
  r.u[0] = *(const uint4*)(p);
  r.u[1] = *(const uint4*)(p + 16);
  return r.v;
}

// B operand (32x16 bf16): lane col = lane&15; lanes 0-15 hold K {0..7, 16..23},
// lanes 16-31 hold K {8..15, 24..31} (same as A). Two b128 loads.
__device__ __forceinline__ v16bf load_opB(const unsigned short* col, int k0, int lane) {
  const unsigned short* p = col + k0 + (((lane >> 4) & 1) << 3);
  BfPack r;
  r.u[0] = *(const uint4*)(p);
  r.u[1] = *(const uint4*)(p + 16);
  return r.v;
}

// One fused layer: D[16,N] = act(A[16,K] @ Wt[N,K]^T + bias), A/Out in LDS.
// Each wave owns two adjacent 16-wide N tiles (shared A load, 2 WMMA chains).
__device__ __forceinline__ void gemm16_bias_gelu_store(
    const unsigned short* __restrict__ A, int strideA,
    const unsigned short* __restrict__ Wt, int K,
    const float* __restrict__ bias, int N,
    unsigned short* __restrict__ Out, int strideOut)
{
  const int lane = threadIdx.x & 31;
  const int wave = threadIdx.x >> 5;
  const int l15  = lane & 15;
  const int mb   = ((lane >> 4) & 1) << 3;         // C/D: VGPR j -> row j+mb
  const unsigned short* arow = A + l15 * strideA;
  for (int n0 = wave * 32; n0 < N; n0 += 256) {
    const int nA = n0 + l15, nB = n0 + 16 + l15;
    const float bvA = bias[nA], bvB = bias[nB];
    v8f accA, accB;
#pragma unroll
    for (int j = 0; j < 8; ++j) { accA[j] = bvA; accB[j] = bvB; }
    const unsigned short* colA = Wt + (size_t)nA * K;
    const unsigned short* colB = Wt + (size_t)nB * K;
    for (int k0 = 0; k0 < K; k0 += 32) {
      v16bf a  = load_opA(arow, k0, lane);
      v16bf bA = load_opB(colA, k0, lane);
      v16bf bB = load_opB(colB, k0, lane);
      accA = __builtin_amdgcn_wmma_f32_16x16x32_bf16(false, a, false, bA, (short)0, accA, false, false);
      accB = __builtin_amdgcn_wmma_f32_16x16x32_bf16(false, a, false, bB, (short)0, accB, false, false);
    }
#pragma unroll
    for (int j = 0; j < 8; ++j) {
      Out[(mb + j) * strideOut + nA] = f32_to_bf16(gelu_exact(accA[j]));
      Out[(mb + j) * strideOut + nB] = f32_to_bf16(gelu_exact(accB[j]));
    }
  }
}

// ---------------------------------------------------------------------------
__global__ void init_kernel(float* __restrict__ out, int* __restrict__ counts) {
  const size_t i = (size_t)blockIdx.x * blockDim.x + threadIdx.x;
  float4 z; z.x = z.y = z.z = z.w = 0.0f;
  ((float4*)out)[i] = z;                      // out_size = 8192*1024, /4 per thread
  if (blockIdx.x == 0 && threadIdx.x < N_EXP) counts[threadIdx.x] = 0;
}

// src:[E][K][N] f32  ->  dst:[E][N][K] bf16   (tiled LDS transpose, RNE cvt)
__global__ __launch_bounds__(256) void cvt_transpose_kernel(
    const float* __restrict__ src, unsigned short* __restrict__ dst, int K, int N)
{
  __shared__ float tile[32][33];
  const size_t eoff = (size_t)blockIdx.z * (size_t)K * (size_t)N;
  const int nb = blockIdx.x * 32;
  const int kb = blockIdx.y * 32;
  const int tx = threadIdx.x;                 // 0..31
  const int ty = threadIdx.y;                 // 0..7
#pragma unroll
  for (int i = 0; i < 32; i += 8)
    tile[ty + i][tx] = src[eoff + (size_t)(kb + ty + i) * N + (nb + tx)];
  __syncthreads();
#pragma unroll
  for (int i = 0; i < 32; i += 8)
    dst[eoff + (size_t)(nb + ty + i) * K + (kb + tx)] = f32_to_bf16(tile[tx][ty + i]);
}

// One wave32 per token: 8 router dot products, top-2 (jax tie-break = lower
// index first, so strict '>' scanning e ascending), append to expert lists.
__global__ __launch_bounds__(256) void router_kernel(
    const float* __restrict__ x, const float* __restrict__ et,
    int* __restrict__ counts, int* __restrict__ list)
{
  const int wave = threadIdx.x >> 5, lane = threadIdx.x & 31;
  const int t = blockIdx.x * 8 + wave;
  const float* xt = x + (size_t)t * D_MODEL;
  float best1 = -__builtin_inff(), best2 = -__builtin_inff();
  int i1 = 0, i2 = 0;
  for (int e = 0; e < N_EXP; ++e) {
    const float* ev = et + e * D_MODEL;
    float acc = 0.0f;
    for (int k = lane; k < D_MODEL; k += 32) acc += xt[k] * ev[k];
#pragma unroll
    for (int m = 16; m >= 1; m >>= 1) acc += __shfl_xor(acc, m, 32);
    if (acc > best1)      { best2 = best1; i2 = i1; best1 = acc; i1 = e; }
    else if (acc > best2) { best2 = acc;   i2 = e; }
  }
  if (lane == 0) {
    int r1 = atomicAdd(&counts[i1], 1); list[i1 * N_TOK + r1] = t;
    int r2 = atomicAdd(&counts[i2], 1); list[i2 * N_TOK + r2] = t;
  }
}

// Fused 3-layer expert MLP: one 16-token tile per workgroup (8 waves).
// LDS: xs 16x1032 bf16 + h1/h2 16x4104 bf16 = 288.75 KB of the 320 KB WGP LDS.
__global__ __launch_bounds__(256, 1) void moe_mlp_kernel(
    const float* __restrict__ x,
    const unsigned short* __restrict__ w1t, const float* __restrict__ b1,
    const unsigned short* __restrict__ w2t, const float* __restrict__ b2,
    const unsigned short* __restrict__ w3t, const float* __restrict__ b3,
    const int* __restrict__ counts, const int* __restrict__ list,
    float* __restrict__ out)
{
  extern __shared__ char smem_raw[];
  unsigned short* xs = (unsigned short*)smem_raw;        // [16][XS_STRIDE]
  unsigned short* h1 = xs + TILE_M * XS_STRIDE;          // [16][H_STRIDE]
  unsigned short* h2 = h1 + TILE_M * H_STRIDE;           // [16][H_STRIDE]
  __shared__ int   toks_s[TILE_M];
  __shared__ float gate_s[TILE_M];

  const int e    = blockIdx.y;
  const int cnt  = counts[e];
  const int base = blockIdx.x * TILE_M;
  if (base >= cnt) return;                               // static grid, early out

  if (threadIdx.x < TILE_M) {
    const int r  = base + (int)threadIdx.x;
    const int ok = (r < cnt);
    toks_s[threadIdx.x] = ok ? list[e * N_TOK + r] : 0;
    gate_s[threadIdx.x] = ok ? 0.5f : 0.0f;              // top-2 => gate 0.5
  }
  __syncthreads();

  for (int i = threadIdx.x; i < TILE_M * D_MODEL; i += 256) {
    const int m = i >> 10, k = i & (D_MODEL - 1);
    xs[m * XS_STRIDE + k] = f32_to_bf16(x[(size_t)toks_s[m] * D_MODEL + k]);
  }
  __syncthreads();

  const unsigned short* W1e = w1t + (size_t)e * D_HID * D_MODEL;
  const unsigned short* W2e = w2t + (size_t)e * D_HID * D_HID;
  const unsigned short* W3e = w3t + (size_t)e * D_MODEL * D_HID;

  gemm16_bias_gelu_store(xs, XS_STRIDE, W1e, D_MODEL, b1 + e * D_HID, D_HID, h1, H_STRIDE);
  __syncthreads();
  gemm16_bias_gelu_store(h1, H_STRIDE, W2e, D_HID, b2 + e * D_HID, D_HID, h2, H_STRIDE);
  __syncthreads();

  // Layer 3: [16,4096] @ [4096,1024] + b3, gate-scaled atomic accumulate.
  const int lane = threadIdx.x & 31;
  const int wave = threadIdx.x >> 5;
  const int l15  = lane & 15;
  const int mb   = ((lane >> 4) & 1) << 3;
  const float* b3e = b3 + e * D_MODEL;
  const unsigned short* arow = h2 + l15 * H_STRIDE;
  for (int n0 = wave * 32; n0 < D_MODEL; n0 += 256) {
    const int nA = n0 + l15, nB = n0 + 16 + l15;
    const float bvA = b3e[nA], bvB = b3e[nB];
    v8f accA, accB;
#pragma unroll
    for (int j = 0; j < 8; ++j) { accA[j] = bvA; accB[j] = bvB; }
    const unsigned short* colA = W3e + (size_t)nA * D_HID;
    const unsigned short* colB = W3e + (size_t)nB * D_HID;
    for (int k0 = 0; k0 < D_HID; k0 += 32) {
      v16bf a  = load_opA(arow, k0, lane);
      v16bf bA = load_opB(colA, k0, lane);
      v16bf bB = load_opB(colB, k0, lane);
      accA = __builtin_amdgcn_wmma_f32_16x16x32_bf16(false, a, false, bA, (short)0, accA, false, false);
      accB = __builtin_amdgcn_wmma_f32_16x16x32_bf16(false, a, false, bB, (short)0, accB, false, false);
    }
#pragma unroll
    for (int j = 0; j < 8; ++j) {
      const int m = mb + j;
      const float g = gate_s[m];
      if (g != 0.0f) {
        float* orow = out + (size_t)toks_s[m] * D_MODEL;
        atomicAdd(orow + nA, g * accA[j]);
        atomicAdd(orow + nB, g * accB[j]);
      }
    }
  }
}

// ---------------------------------------------------------------------------
extern "C" void kernel_launch(void* const* d_in, const int* in_sizes, int n_in,
                              void* d_out, int out_size, void* d_ws, size_t ws_size,
                              hipStream_t stream) {
  (void)in_sizes; (void)n_in; (void)out_size; (void)ws_size;
  const float* x  = (const float*)d_in[0];
  const float* et = (const float*)d_in[1];
  const float* W1 = (const float*)d_in[2];
  const float* b1 = (const float*)d_in[3];
  const float* W2 = (const float*)d_in[4];
  const float* b2 = (const float*)d_in[5];
  const float* W3 = (const float*)d_in[6];
  const float* b3 = (const float*)d_in[7];
  float* out = (float*)d_out;

  char* ws = (char*)d_ws;
  int* counts = (int*)ws;                                       // 32B (pad 256)
  int* list   = (int*)(ws + 256);                               // 8*8192*4
  unsigned short* w1t = (unsigned short*)(ws + 256 + (size_t)N_EXP * N_TOK * 4);
  unsigned short* w2t = w1t + (size_t)N_EXP * D_HID * D_MODEL;  // 64MB after w1t
  unsigned short* w3t = w2t + (size_t)N_EXP * D_HID * D_HID;    // +256MB, +64MB

  init_kernel<<<(N_TOK * D_MODEL) / (256 * 4), 256, 0, stream>>>(out, counts);
  cvt_transpose_kernel<<<dim3(D_HID / 32, D_MODEL / 32, N_EXP), dim3(32, 8), 0, stream>>>(W1, w1t, D_MODEL, D_HID);
  cvt_transpose_kernel<<<dim3(D_HID / 32, D_HID / 32, N_EXP), dim3(32, 8), 0, stream>>>(W2, w2t, D_HID, D_HID);
  cvt_transpose_kernel<<<dim3(D_MODEL / 32, D_HID / 32, N_EXP), dim3(32, 8), 0, stream>>>(W3, w3t, D_HID, D_MODEL);
  router_kernel<<<N_TOK / 8, 256, 0, stream>>>(x, et, counts, list);

  const size_t smem = (size_t)2 * (TILE_M * XS_STRIDE + 2 * TILE_M * H_STRIDE); // 295680B
  hipFuncSetAttribute((const void*)moe_mlp_kernel,
                      hipFuncAttributeMaxDynamicSharedMemorySize, (int)smem);
  moe_mlp_kernel<<<dim3(N_TOK / TILE_M, N_EXP), 256, smem, stream>>>(
      x, w1t, b1, w2t, b2, w3t, b3, counts, list, out);
}
